// MoEAttention_84928683311111
// MI455X (gfx1250) — matrix operand order repaired and checked by reading, not compiled, not run on hardware
//
#include <hip/hip_runtime.h>
#include <hip/hip_bf16.h>

// ---------------------------------------------------------------------------
// MoE attention for MI455X (gfx1250): bf16 WMMA pipeline, f32 accumulation.
// Stages: convert -> router -> per-expert QKV GEMM -> flash attention -> out GEMM
// GEMMs use 64x64 register-blocked tiles (4x4 WMMA per wave) for 4x data reuse.
// ---------------------------------------------------------------------------

#define D_MODEL   1280
#define NUM_EXP   20
#define HEAD_DIM  64
#define QKV_N     192          // 3*HEAD_DIM
#define BATCH     4
#define SEQ       1024
#define ROWS      (BATCH * SEQ)    // 4096
#define ATT_SCALE 0.125f           // 64^-0.5

typedef __attribute__((ext_vector_type(16))) __bf16 v16bf;
typedef __attribute__((ext_vector_type(8)))  float  v8f;

union FragU { v16bf v; uint4 u[2]; };

__device__ __forceinline__ unsigned short f2bf(float f) {
    unsigned u = __builtin_bit_cast(unsigned, f);
    u += 0x7FFFu + ((u >> 16) & 1u);          // round-to-nearest-even
    return (unsigned short)(u >> 16);
}

// A fragment (16x32 bf16, row-major source, row pointer given): lane m fixed,
// elements [k0 + half*8 .. +7] and [k0 + 16 + half*8 .. +7]  (2x 16B loads)
__device__ __forceinline__ v16bf load_a(const unsigned short* row, int k0, int half) {
    const unsigned short* p = row + k0 + half * 8;
    FragU f;
    f.u[0] = *(const uint4*)(p);
    f.u[1] = *(const uint4*)(p + 16);
    return f.v;
}

// B fragment (32x16 bf16) from a K-contiguous ("transposed") source row:
// lane n fixed, elements [k0 + half*8 .. +7] and [k0 + 16 + half*8 .. +7]  (2x 16B loads)
__device__ __forceinline__ v16bf load_b(const unsigned short* row, int k0, int half) {
    const unsigned short* p = row + k0 + half * 8;
    FragU f;
    f.u[0] = *(const uint4*)(p);
    f.u[1] = *(const uint4*)(p + 16);
    return f.v;
}

__device__ __forceinline__ v8f wmma_bf16(v16bf a, v16bf b, v8f c) {
    return __builtin_amdgcn_wmma_f32_16x16x32_bf16(
        false, a, false, b, (short)0, c, false, false);
}

// ---------------------------------------------------------------------------
// Stage 0: conversions / transposes to bf16
// ---------------------------------------------------------------------------
__global__ void cvt_x_kernel(const float* __restrict__ x, unsigned short* __restrict__ xb, int n) {
    int i = blockIdx.x * blockDim.x + threadIdx.x;
    if (i < n) xb[i] = f2bf(x[i]);
}

// wqkv [E][D][192] -> wqkvT [E][192][D] (bf16)
__global__ void cvt_wqkv_kernel(const float* __restrict__ w, unsigned short* __restrict__ wT) {
    int i = blockIdx.x * blockDim.x + threadIdx.x;   // dest-major index
    int k = i % D_MODEL;
    int n = (i / D_MODEL) % QKV_N;
    int e = i / (D_MODEL * QKV_N);
    wT[i] = f2bf(w[((size_t)e * D_MODEL + k) * QKV_N + n]);
}

// out_w [D][D] -> out_wT [D][D] transposed (bf16)
__global__ void cvt_outw_kernel(const float* __restrict__ w, unsigned short* __restrict__ wT) {
    int i = blockIdx.x * blockDim.x + threadIdx.x;   // n*D + k
    int k = i % D_MODEL;
    int n = i / D_MODEL;
    wT[i] = f2bf(w[(size_t)k * D_MODEL + n]);
}

// ---------------------------------------------------------------------------
// Stage 1: router softmax gates [4096][20]  (tiny, scalar)
// ---------------------------------------------------------------------------
__global__ void router_kernel(const float* __restrict__ x, const float* __restrict__ rw,
                              const float* __restrict__ rb, float* __restrict__ gates) {
    int row = blockIdx.x * blockDim.x + threadIdx.x;
    if (row >= ROWS) return;
    float acc[NUM_EXP];
#pragma unroll
    for (int e = 0; e < NUM_EXP; e++) acc[e] = rb[e];
    const float* xr = x + (size_t)row * D_MODEL;
    for (int k = 0; k < D_MODEL; k++) {
        float xv = xr[k];
        const float* wr = rw + (size_t)k * NUM_EXP;
#pragma unroll
        for (int e = 0; e < NUM_EXP; e++) acc[e] = fmaf(xv, wr[e], acc[e]);
    }
    float mx = acc[0];
#pragma unroll
    for (int e = 1; e < NUM_EXP; e++) mx = fmaxf(mx, acc[e]);
    float s = 0.f;
#pragma unroll
    for (int e = 0; e < NUM_EXP; e++) { acc[e] = __expf(acc[e] - mx); s += acc[e]; }
    float inv = 1.f / s;
#pragma unroll
    for (int e = 0; e < NUM_EXP; e++) gates[(size_t)row * NUM_EXP + e] = acc[e] * inv;
}

// ---------------------------------------------------------------------------
// Stage 2: per-expert QKV GEMM: [4096x1280] @ [1280x192] + bias -> q,k,vT (bf16)
// one wave per 64x64 block (4x4 WMMA tiles); blocks = E(20) * 64 * 3
// nblk==0 -> Q slice, nblk==1 -> K slice, nblk==2 -> V slice (wave-uniform)
// ---------------------------------------------------------------------------
__global__ __launch_bounds__(256) void qkv_gemm_kernel(
    const unsigned short* __restrict__ xb, const unsigned short* __restrict__ wT,
    const float* __restrict__ bqkv, unsigned short* __restrict__ qb,
    unsigned short* __restrict__ kb, unsigned short* __restrict__ vT) {
    int gw   = blockIdx.x * 8 + (threadIdx.x >> 5);
    int lane = threadIdx.x & 31;
    int nblk = gw % 3;
    int mblk = (gw / 3) % 64;
    int e    = gw / (3 * 64);
    int m = lane & 15, half = lane >> 4;

    const unsigned short* Arow[4];
    const unsigned short* Brow[4];
#pragma unroll
    for (int i = 0; i < 4; i++)
        Arow[i] = xb + (size_t)(mblk * 64 + i * 16 + m) * D_MODEL;
#pragma unroll
    for (int j = 0; j < 4; j++)
        Brow[j] = wT + ((size_t)e * QKV_N + nblk * 64 + j * 16 + m) * D_MODEL;

    v8f acc[4][4] = {};
    for (int k0 = 0; k0 < D_MODEL; k0 += 32) {
        v16bf a[4], b[4];
#pragma unroll
        for (int i = 0; i < 4; i++) {
            __builtin_prefetch(Arow[i] + k0 + 128, 0, 1);
            a[i] = load_a(Arow[i], k0, half);
        }
#pragma unroll
        for (int j = 0; j < 4; j++) {
            __builtin_prefetch(Brow[j] + k0 + 128, 0, 1);
            b[j] = load_b(Brow[j], k0, half);
        }
#pragma unroll
        for (int i = 0; i < 4; i++)
#pragma unroll
            for (int j = 0; j < 4; j++)
                acc[i][j] = wmma_bf16(a[i], b[j], acc[i][j]);
    }

    // epilogue: bias add, bf16 convert, scatter to q/k/vT (wave-uniform by nblk)
#pragma unroll
    for (int j = 0; j < 4; j++) {
        int ncol = nblk * 64 + j * 16 + m;    // column in [0,192)
        int c64  = ncol & 63;                 // column within q/k/v slice
        float bias = bqkv[e * QKV_N + ncol];
#pragma unroll
        for (int i = 0; i < 4; i++) {
#pragma unroll
            for (int r = 0; r < 8; r++) {
                int row = mblk * 64 + i * 16 + r + 8 * half;   // = b*1024 + s
                unsigned short bv = f2bf(acc[i][j][r] + bias);
                if (nblk == 0) {
                    qb[((size_t)e * ROWS + row) * HEAD_DIM + c64] = bv;
                } else if (nblk == 1) {
                    kb[((size_t)e * ROWS + row) * HEAD_DIM + c64] = bv;
                } else {
                    int b = row >> 10, s = row & 1023;
                    vT[(((size_t)e * BATCH + b) * HEAD_DIM + c64) * SEQ + s] = bv;
                }
            }
        }
    }
}

// ---------------------------------------------------------------------------
// Stage 3: flash attention + gate. One wave per 16-row Q tile.
// tiles = E(20) * B(4) * Qtiles(64) = 5120
// ---------------------------------------------------------------------------
__global__ __launch_bounds__(256) void attn_kernel(
    const unsigned short* __restrict__ qb, const unsigned short* __restrict__ kb,
    const unsigned short* __restrict__ vT, const float* __restrict__ gates,
    unsigned short* __restrict__ comb) {
    __shared__ unsigned short Plds[8 * 16 * 32];   // per-wave 16x32 P tile (bf16)
    int wid = threadIdx.x >> 5, lane = threadIdx.x & 31;
    int gw = blockIdx.x * 8 + wid;
    int qtile = gw % 64;
    int b     = (gw / 64) % BATCH;
    int e     = gw / (64 * BATCH);
    int m = lane & 15, half = lane >> 4;

    const unsigned short* Q = qb + ((size_t)e * ROWS + b * SEQ) * HEAD_DIM;
    const unsigned short* K = kb + ((size_t)e * ROWS + b * SEQ) * HEAD_DIM;
    const unsigned short* V = vT + ((size_t)e * BATCH + b) * HEAD_DIM * SEQ;

    const unsigned short* qrow = Q + (size_t)(qtile * 16 + m) * HEAD_DIM;
    v16bf aq0 = load_a(qrow, 0, half);
    v16bf aq1 = load_a(qrow, 32, half);

    v8f acc0 = {}, acc1 = {}, acc2 = {}, acc3 = {};
    float rmax[8], rsum[8];
#pragma unroll
    for (int r = 0; r < 8; r++) { rmax[r] = -1e30f; rsum[r] = 0.f; }

    unsigned short* Pw = Plds + wid * 512;

    for (int kc = 0; kc < SEQ / 32; kc++) {
        int keybase = kc * 32;
        // ---- scores: 16 q-rows x 32 keys (two 16x16 tiles, K=64 over h) ----
        const unsigned short* krow0 = K + (size_t)(keybase + m) * HEAD_DIM;
        const unsigned short* krow1 = K + (size_t)(keybase + 16 + m) * HEAD_DIM;
        v8f s0 = {}, s1 = {};
        s0 = wmma_bf16(aq0, load_b(krow0, 0, half), s0);
        s0 = wmma_bf16(aq1, load_b(krow0, 32, half), s0);
        s1 = wmma_bf16(aq0, load_b(krow1, 0, half), s1);
        s1 = wmma_bf16(aq1, load_b(krow1, 32, half), s1);

        // ---- online softmax over this 32-key chunk (per row reductions) ----
#pragma unroll
        for (int r = 0; r < 8; r++) {
            float x0 = s0[r] * ATT_SCALE, x1 = s1[r] * ATT_SCALE;
            float cmax = fmaxf(x0, x1);
            cmax = fmaxf(cmax, __shfl_xor(cmax, 1));
            cmax = fmaxf(cmax, __shfl_xor(cmax, 2));
            cmax = fmaxf(cmax, __shfl_xor(cmax, 4));
            cmax = fmaxf(cmax, __shfl_xor(cmax, 8));
            float mnew  = fmaxf(rmax[r], cmax);
            float alpha = __expf(rmax[r] - mnew);
            float p0 = __expf(x0 - mnew), p1 = __expf(x1 - mnew);
            float psum = p0 + p1;
            psum += __shfl_xor(psum, 1);
            psum += __shfl_xor(psum, 2);
            psum += __shfl_xor(psum, 4);
            psum += __shfl_xor(psum, 8);
            rsum[r] = rsum[r] * alpha + psum;
            rmax[r] = mnew;
            acc0[r] *= alpha; acc1[r] *= alpha; acc2[r] *= alpha; acc3[r] *= alpha;
            int row = r + 8 * half;                 // C-layout row this lane holds
            Pw[row * 32 + m]      = f2bf(p0);       // key = m      (tile 0)
            Pw[row * 32 + 16 + m] = f2bf(p1);       // key = 16 + m (tile 1)
        }
        // stores and loads below both target LDS of this wave; DS ops are
        // in-order per wave — make the stores visibly complete first.
        asm volatile("s_wait_dscnt 0x0" ::: "memory");

        // ---- redistribute P (C layout -> A layout) via LDS ----
        v16bf ap = load_a(Pw + m * 32, 0, half);    // 16x32 P tile, row stride 32

        // ---- acc += P @ V  (4 output n-tiles of 16 h-columns) ----
        const unsigned short* vbase = V + (size_t)m * SEQ + keybase;
        acc0 = wmma_bf16(ap, load_b(vbase,            0, half), acc0);
        acc1 = wmma_bf16(ap, load_b(vbase + 16 * SEQ, 0, half), acc1);
        acc2 = wmma_bf16(ap, load_b(vbase + 32 * SEQ, 0, half), acc2);
        acc3 = wmma_bf16(ap, load_b(vbase + 48 * SEQ, 0, half), acc3);
        asm volatile("" ::: "memory");              // keep next-iter P stores after loads
    }

    // ---- epilogue: normalize, gate, write bf16 combined [4096][1280] ----
#pragma unroll
    for (int r = 0; r < 8; r++) {
        int s = qtile * 16 + r + 8 * half;
        size_t grow = (size_t)(b * SEQ + s);
        float g   = gates[grow * NUM_EXP + e];
        float inv = g / rsum[r];
        unsigned short* crow = comb + grow * D_MODEL + e * HEAD_DIM + m;
        crow[0]  = f2bf(acc0[r] * inv);
        crow[16] = f2bf(acc1[r] * inv);
        crow[32] = f2bf(acc2[r] * inv);
        crow[48] = f2bf(acc3[r] * inv);
    }
}

// ---------------------------------------------------------------------------
// Stage 4: output projection: [4096x1280] @ [1280x1280] + bias -> f32 out
// one wave per 64x64 block (4x4 WMMA tiles); blocks = 64 * 20
// ---------------------------------------------------------------------------
__global__ __launch_bounds__(256) void out_gemm_kernel(
    const unsigned short* __restrict__ comb, const unsigned short* __restrict__ wT,
    const float* __restrict__ outb, float* __restrict__ out) {
    int gw   = blockIdx.x * 8 + (threadIdx.x >> 5);
    int lane = threadIdx.x & 31;
    int nblk = gw % 20;
    int mblk = gw / 20;
    int m = lane & 15, half = lane >> 4;

    const unsigned short* Arow[4];
    const unsigned short* Brow[4];
#pragma unroll
    for (int i = 0; i < 4; i++)
        Arow[i] = comb + (size_t)(mblk * 64 + i * 16 + m) * D_MODEL;
#pragma unroll
    for (int j = 0; j < 4; j++)
        Brow[j] = wT + (size_t)(nblk * 64 + j * 16 + m) * D_MODEL;

    v8f acc[4][4] = {};
    for (int k0 = 0; k0 < D_MODEL; k0 += 32) {
        v16bf a[4], b[4];
#pragma unroll
        for (int i = 0; i < 4; i++) {
            __builtin_prefetch(Arow[i] + k0 + 128, 0, 1);
            a[i] = load_a(Arow[i], k0, half);
        }
#pragma unroll
        for (int j = 0; j < 4; j++) {
            __builtin_prefetch(Brow[j] + k0 + 128, 0, 1);
            b[j] = load_b(Brow[j], k0, half);
        }
#pragma unroll
        for (int i = 0; i < 4; i++)
#pragma unroll
            for (int j = 0; j < 4; j++)
                acc[i][j] = wmma_bf16(a[i], b[j], acc[i][j]);
    }

#pragma unroll
    for (int j = 0; j < 4; j++) {
        int ncol = nblk * 64 + j * 16 + m;
        float bias = outb[ncol];
#pragma unroll
        for (int i = 0; i < 4; i++) {
#pragma unroll
            for (int r = 0; r < 8; r++) {
                int row = mblk * 64 + i * 16 + r + 8 * half;
                out[(size_t)row * D_MODEL + ncol] = acc[i][j][r] + bias;
            }
        }
    }
}

// ---------------------------------------------------------------------------
// Host launcher
// ---------------------------------------------------------------------------
extern "C" void kernel_launch(void* const* d_in, const int* in_sizes, int n_in,
                              void* d_out, int out_size, void* d_ws, size_t ws_size,
                              hipStream_t stream) {
    const float* x        = (const float*)d_in[0];
    const float* wqkv     = (const float*)d_in[1];
    const float* bqkv     = (const float*)d_in[2];
    const float* router_w = (const float*)d_in[3];
    const float* router_b = (const float*)d_in[4];
    const float* out_w    = (const float*)d_in[5];
    const float* out_b    = (const float*)d_in[6];
    float* out = (float*)d_out;

    char* ws = (char*)d_ws;
    size_t off = 0;
    auto alloc = [&](size_t bytes) { char* p = ws + off; off += (bytes + 255) & ~(size_t)255; return p; };

    unsigned short* xb     = (unsigned short*)alloc((size_t)ROWS * D_MODEL * 2);
    unsigned short* wqkvT  = (unsigned short*)alloc((size_t)NUM_EXP * QKV_N * D_MODEL * 2);
    unsigned short* outwT  = (unsigned short*)alloc((size_t)D_MODEL * D_MODEL * 2);
    float*          gates  = (float*)        alloc((size_t)ROWS * NUM_EXP * 4);
    unsigned short* qb     = (unsigned short*)alloc((size_t)NUM_EXP * ROWS * HEAD_DIM * 2);
    unsigned short* kb     = (unsigned short*)alloc((size_t)NUM_EXP * ROWS * HEAD_DIM * 2);
    unsigned short* vT     = (unsigned short*)alloc((size_t)NUM_EXP * ROWS * HEAD_DIM * 2);
    unsigned short* comb   = (unsigned short*)alloc((size_t)ROWS * D_MODEL * 2);
    (void)ws_size; (void)n_in; (void)in_sizes; (void)out_size;

    // Stage 0: conversions
    {
        int n = ROWS * D_MODEL;                                 // 5,242,880
        cvt_x_kernel<<<n / 256, 256, 0, stream>>>(x, xb, n);
    }
    cvt_wqkv_kernel<<<(NUM_EXP * QKV_N * D_MODEL) / 256, 256, 0, stream>>>(wqkv, wqkvT);
    cvt_outw_kernel<<<(D_MODEL * D_MODEL) / 256, 256, 0, stream>>>(out_w, outwT);

    // Stage 1: router
    router_kernel<<<ROWS / 256, 256, 0, stream>>>(x, router_w, router_b, gates);

    // Stage 2: per-expert QKV GEMM  (20 * 64 * 3 blocks of 64x64 / 8 waves)
    qkv_gemm_kernel<<<(NUM_EXP * 64 * 3) / 8, 256, 0, stream>>>(xb, wqkvT, bqkv, qb, kb, vT);

    // Stage 3: flash attention + gating  (20*4*64 q-tiles / 8 waves)
    attn_kernel<<<(NUM_EXP * BATCH * 64) / 8, 256, 0, stream>>>(qb, kb, vT, gates, comb);

    // Stage 4: output projection  (64 * 20 blocks of 64x64 / 8 waves)
    out_gemm_kernel<<<(64 * 20) / 8, 256, 0, stream>>>(comb, outwT, out_b, out);
}
